// AttentionSeq2Seq_73177652789315
// MI455X (gfx1250) — matrix operand order repaired and checked by reading, not compiled, not run on hardware
//
#include <hip/hip_runtime.h>
#include <stdint.h>

// ---------------- problem constants ----------------
static constexpr int TT = 128;   // time steps (enc and dec)
static constexpr int NN = 128;   // batch
static constexpr int CC = 512;   // input channels
static constexpr int DD = 1024;  // hidden
static constexpr int KK = 512;   // output classes

// ---------------- WMMA types ----------------
typedef __attribute__((ext_vector_type(16))) __bf16   v16bf;
typedef __attribute__((ext_vector_type(8)))  float    v8f;
typedef __attribute__((ext_vector_type(4)))  unsigned u32x4;

union Frag { v16bf v; u32x4 q[2]; };

__device__ __forceinline__ uint16_t f2b(float f) {
  uint32_t u = __float_as_uint(f);
  uint32_t r = u + 0x7FFFu + ((u >> 16) & 1u);   // round-to-nearest-even
  return (uint16_t)(r >> 16);
}
__device__ __forceinline__ float b2f(uint16_t b) {
  return __uint_as_float(((uint32_t)b) << 16);
}

// 32x32 macro-tile (2x2 register blocking): 4 accumulators per wave.
// Per K-step of 32: 8 global_load_b128 feed 4 v_wmma_f32_16x16x32_bf16
// (2 loads per WMMA instead of 4 for an unblocked tile).
// A : row-major bf16 [32 x Ktot], row r at A + r*lda (lda multiple of 8)
// Bt: transposed weights, row n (output column) at Bt + n*ldb, 32 columns
__device__ __forceinline__ void gemm_macro_acc(v8f acc[2][2],
    const uint16_t* __restrict__ A, int lda,
    const uint16_t* __restrict__ Bt, int ldb, int Ktot)
{
  const int lane = threadIdx.x & 31;
  const int r    = lane & 15;
  const int hi   = lane >> 4;
  const uint16_t* ap0 = A  + (size_t)r * lda + 8  * hi;  // A frag rows 0..15
  const uint16_t* ap1 = ap0 + (size_t)16 * lda;           // A frag rows 16..31
  const uint16_t* bp0 = Bt + (size_t)r * ldb + 8  * hi;  // B frag cols 0..15
  const uint16_t* bp1 = bp0 + (size_t)16 * ldb;           // B frag cols 16..31
  for (int k0 = 0; k0 < Ktot; k0 += 32) {
    Frag fa0, fa1, fb0, fb1;
    fa0.q[0] = *reinterpret_cast<const u32x4*>(ap0 + k0);
    fa0.q[1] = *reinterpret_cast<const u32x4*>(ap0 + k0 + 16);
    fa1.q[0] = *reinterpret_cast<const u32x4*>(ap1 + k0);
    fa1.q[1] = *reinterpret_cast<const u32x4*>(ap1 + k0 + 16);
    fb0.q[0] = *reinterpret_cast<const u32x4*>(bp0 + k0);
    fb0.q[1] = *reinterpret_cast<const u32x4*>(bp0 + k0 + 16);
    fb1.q[0] = *reinterpret_cast<const u32x4*>(bp1 + k0);
    fb1.q[1] = *reinterpret_cast<const u32x4*>(bp1 + k0 + 16);
    acc[0][0] = __builtin_amdgcn_wmma_f32_16x16x32_bf16(
        false, fa0.v, false, fb0.v, (short)0, acc[0][0], false, false);
    acc[0][1] = __builtin_amdgcn_wmma_f32_16x16x32_bf16(
        false, fa0.v, false, fb1.v, (short)0, acc[0][1], false, false);
    acc[1][0] = __builtin_amdgcn_wmma_f32_16x16x32_bf16(
        false, fa1.v, false, fb0.v, (short)0, acc[1][0], false, false);
    acc[1][1] = __builtin_amdgcn_wmma_f32_16x16x32_bf16(
        false, fa1.v, false, fb1.v, (short)0, acc[1][1], false, false);
  }
}

// ---------------- conversion / transpose kernels ----------------
__global__ void att_convert_bf(const float* __restrict__ in,
                               uint16_t* __restrict__ out, int count) {
  int i = blockIdx.x * blockDim.x + threadIdx.x;
  if (i < count) out[i] = f2b(in[i]);
}

// in: f32 [R x Cc] row-major  ->  out: bf16 [Cc x R] row-major (transposed)
__global__ void att_transpose_bf(const float* __restrict__ in,
                                 uint16_t* __restrict__ out, int R, int Cc) {
  int i = blockIdx.x * blockDim.x + threadIdx.x;
  if (i >= R * Cc) return;
  int r = i / Cc, c = i % Cc;
  out[(size_t)c * R + r] = f2b(in[i]);
}

__global__ void att_init_idx(int* __restrict__ idx) {
  idx[threadIdx.x] = CC - 2;   // one-hot start token s[:, -2] = 1
}

// ---------------- generic GEMM (optional bias, f32 and/or bf16 out) --------
// macro tiles of 32x32; ntilesN = Ncols/32
__global__ void att_gemm(const uint16_t* __restrict__ A, int lda,
                         const uint16_t* __restrict__ Bt, int ldb, int Ktot,
                         const float* __restrict__ bias,
                         float* __restrict__ outF, uint16_t* __restrict__ outB,
                         int ldc, int ntilesN) {
  int tile = blockIdx.x * blockDim.y + threadIdx.y;
  int tm = tile / ntilesN, tn = tile % ntilesN;
  int m0 = tm * 32, n0 = tn * 32;
  v8f acc[2][2] = {};
  gemm_macro_acc(acc, A + (size_t)m0 * lda, lda, Bt + (size_t)n0 * ldb, ldb, Ktot);
  int lane = threadIdx.x & 31;
#pragma unroll
  for (int mi = 0; mi < 2; mi++) {
#pragma unroll
    for (int ni = 0; ni < 2; ni++) {
      int col = n0 + ni * 16 + (lane & 15);
      int rb  = m0 + mi * 16 + 8 * (lane >> 4);
      float bcol = bias ? bias[col] : 0.f;
#pragma unroll
      for (int v = 0; v < 8; v++) {
        float val = acc[mi][ni][v] + bcol;
        size_t o = (size_t)(rb + v) * ldc + col;
        if (outF) outF[o] = val;
        if (outB) outB[o] = f2b(val);
      }
    }
  }
}

// ---------------- encoder step: h = relu(x_t@Wx + bx + h@Wh + bh) ----------
__global__ void att_enc_step(const uint16_t* __restrict__ xt,    // [N x C] bf16
                             const uint16_t* __restrict__ hprev, // [N x D] bf16
                             int use_h,
                             const uint16_t* __restrict__ WxT,   // [D x C]
                             const uint16_t* __restrict__ WhT,   // [D x D]
                             const float* __restrict__ bx,
                             const float* __restrict__ bh,
                             float* __restrict__ hf,
                             uint16_t* __restrict__ hbf) {
  int tile = blockIdx.x * blockDim.y + threadIdx.y;   // 4 x 32 macro tiles
  int tm = tile >> 5, tn = tile & 31;
  int m0 = tm * 32, n0 = tn * 32;
  v8f acc[2][2] = {};
  gemm_macro_acc(acc, xt + (size_t)m0 * CC, CC, WxT + (size_t)n0 * CC, CC, CC);
  if (use_h)
    gemm_macro_acc(acc, hprev + (size_t)m0 * DD, DD, WhT + (size_t)n0 * DD, DD, DD);
  int lane = threadIdx.x & 31;
#pragma unroll
  for (int mi = 0; mi < 2; mi++) {
#pragma unroll
    for (int ni = 0; ni < 2; ni++) {
      int col = n0 + ni * 16 + (lane & 15);
      int rb  = m0 + mi * 16 + 8 * (lane >> 4);
      float badd = bx[col] + bh[col];
#pragma unroll
      for (int v = 0; v < 8; v++) {
        float val = fmaxf(acc[mi][ni][v] + badd, 0.f);
        size_t o = (size_t)(rb + v) * DD + col;
        hf[o] = val;
        hbf[o] = f2b(val);
      }
    }
  }
}

// ---- decoder h step: h = relu(dec_Wx[idx[n],:] + bx + h@Wh + bh) ----------
__global__ void att_dec_hstep(const uint16_t* __restrict__ hprev, // [N x D] bf16
                              const uint16_t* __restrict__ WhT,   // [D x D]
                              const float* __restrict__ Wx_rows,  // dec_Wx f32 [C x D]
                              const int* __restrict__ sidx,
                              const float* __restrict__ bx,
                              const float* __restrict__ bh,
                              float* __restrict__ hf,
                              uint16_t* __restrict__ hbf) {
  int tile = blockIdx.x * blockDim.y + threadIdx.y;   // 4 x 32 macro tiles
  int tm = tile >> 5, tn = tile & 31;
  int m0 = tm * 32, n0 = tn * 32;
  v8f acc[2][2] = {};
  gemm_macro_acc(acc, hprev + (size_t)m0 * DD, DD, WhT + (size_t)n0 * DD, DD, DD);
  int lane = threadIdx.x & 31;
#pragma unroll
  for (int mi = 0; mi < 2; mi++) {
#pragma unroll
    for (int ni = 0; ni < 2; ni++) {
      int col = n0 + ni * 16 + (lane & 15);
      int rb  = m0 + mi * 16 + 8 * (lane >> 4);
      float badd = bx[col] + bh[col];
#pragma unroll
      for (int v = 0; v < 8; v++) {
        int row = rb + v;
        float g = Wx_rows[(size_t)sidx[row] * DD + col];  // one-hot row gather
        float val = fmaxf(acc[mi][ni][v] + g + badd, 0.f);
        size_t o = (size_t)row * DD + col;
        hf[o] = val;
        hbf[o] = f2b(val);
      }
    }
  }
}

// ---- y = y_rnn @ Wc[:K] + c @ Wc[K:] + bc  (fused concat GEMM) ------------
__global__ void att_dec_y(const uint16_t* __restrict__ yrnn, // [N x K] bf16
                          const uint16_t* __restrict__ cbf,  // [N x D] bf16
                          const uint16_t* __restrict__ WcT,  // [K x (K+D)]
                          const float* __restrict__ bc,
                          float* __restrict__ yout) {        // [N x K] f32
  int tile = blockIdx.x * blockDim.y + threadIdx.y;   // 4 x 16 macro tiles
  int tm = tile >> 4, tn = tile & 15;
  int m0 = tm * 32, n0 = tn * 32;
  const int LDW = KK + DD;  // 1536
  v8f acc[2][2] = {};
  gemm_macro_acc(acc, yrnn + (size_t)m0 * KK, KK, WcT + (size_t)n0 * LDW,      LDW, KK);
  gemm_macro_acc(acc, cbf  + (size_t)m0 * DD, DD, WcT + (size_t)n0 * LDW + KK, LDW, DD);
  int lane = threadIdx.x & 31;
#pragma unroll
  for (int mi = 0; mi < 2; mi++) {
#pragma unroll
    for (int ni = 0; ni < 2; ni++) {
      int col = n0 + ni * 16 + (lane & 15);
      int rb  = m0 + mi * 16 + 8 * (lane >> 4);
      float bcol = bc[col];
#pragma unroll
      for (int v = 0; v < 8; v++)
        yout[(size_t)(rb + v) * KK + col] = acc[mi][ni][v] + bcol;
    }
  }
}

// ---- attention scores e[t,n] = <pre[t,n,:], h[n,:]>  (wave per output) ----
__global__ void att_edot(const uint16_t* __restrict__ pre_bf,
                         const float* __restrict__ h,
                         float* __restrict__ e) {
  int w = blockIdx.x * (blockDim.x >> 5) + (threadIdx.x >> 5);
  int t = w >> 7, n = w & 127;
  int lane = threadIdx.x & 31;
  const uint16_t* p = pre_bf + ((size_t)t * NN + n) * DD;
  const float* hh = h + (size_t)n * DD;
  float s = 0.f;
  for (int d = lane; d < DD; d += 32) s += b2f(p[d]) * hh[d];
#pragma unroll
  for (int off = 16; off; off >>= 1) s += __shfl_xor(s, off, 32);
  if (lane == 0) e[t * NN + n] = s;
}

// ---- softmax over encoder time (block per n) ------------------------------
__global__ void att_softmax(const float* __restrict__ e,
                            float* __restrict__ a,
                            float* __restrict__ aij_slice) {   // + t_dec*T*N
  int n = blockIdx.x, t = threadIdx.x;
  __shared__ float sm[TT];
  float v = e[t * NN + n];
  sm[t] = v; __syncthreads();
  for (int s = 64; s; s >>= 1) { if (t < s) sm[t] = fmaxf(sm[t], sm[t + s]); __syncthreads(); }
  float mx = sm[0]; __syncthreads();
  float ex = __expf(v - mx);
  sm[t] = ex; __syncthreads();
  for (int s = 64; s; s >>= 1) { if (t < s) sm[t] += sm[t + s]; __syncthreads(); }
  float av = ex / sm[0];
  a[t * NN + n] = av;
  aij_slice[t * NN + n] = av;
}

// ---- context c[n,:] = sum_t a[t,n] * enc_h[t,n,:] -------------------------
__global__ void att_ctx(const float* __restrict__ a,
                        const float* __restrict__ enc_h,
                        uint16_t* __restrict__ cbf) {
  int i = blockIdx.x * blockDim.x + threadIdx.x;  // n*D + d
  int n = i >> 10, d = i & 1023;
  float s = 0.f;
  for (int t = 0; t < TT; t++)
    s += a[t * NN + n] * enc_h[((size_t)t * NN + n) * DD + d];
  cbf[i] = f2b(s);
}

// ---- argmax over K per row (first max wins, like jnp.argmax) --------------
__global__ void att_argmax(const float* __restrict__ y, int* __restrict__ idx) {
  int n = blockIdx.x, t = threadIdx.x;
  __shared__ float bv[128]; __shared__ int bi[128];
  float best = -3.4e38f; int besti = 0;
  for (int k = t; k < KK; k += 128) {
    float v = y[(size_t)n * KK + k];
    if (v > best) { best = v; besti = k; }
  }
  bv[t] = best; bi[t] = besti; __syncthreads();
  for (int s = 64; s; s >>= 1) {
    if (t < s) {
      if (bv[t + s] > bv[t] || (bv[t + s] == bv[t] && bi[t + s] < bi[t])) {
        bv[t] = bv[t + s]; bi[t] = bi[t + s];
      }
    }
    __syncthreads();
  }
  if (t == 0) idx[n] = bi[0];
}

// ---------------- host orchestration ----------------
extern "C" void kernel_launch(void* const* d_in, const int* in_sizes, int n_in,
                              void* d_out, int out_size, void* d_ws, size_t ws_size,
                              hipStream_t stream) {
  const float* x       = (const float*)d_in[0];
  const float* enc_Wx  = (const float*)d_in[1];
  const float* enc_bx  = (const float*)d_in[2];
  const float* enc_Wh  = (const float*)d_in[3];
  const float* enc_bh  = (const float*)d_in[4];
  const float* dec_Wx  = (const float*)d_in[5];
  const float* dec_bx  = (const float*)d_in[6];
  const float* dec_Wh  = (const float*)d_in[7];
  const float* dec_bh  = (const float*)d_in[8];
  const float* dec_Wy  = (const float*)d_in[9];
  const float* dec_by  = (const float*)d_in[10];
  const float* W_alpha = (const float*)d_in[11];
  const float* Wc      = (const float*)d_in[12];
  const float* bc      = (const float*)d_in[13];

  float* yout = (float*)d_out;                       // (T, N, K) f32
  float* aij  = yout + (size_t)TT * NN * KK;         // (T, T, N) f32

  // workspace carve-out (256B aligned)
  char* cur = (char*)d_ws;
  auto carve = [&](size_t bytes) -> void* {
    void* p = cur; cur += (bytes + 255) & ~(size_t)255; return p;
  };
  uint16_t* x_bf    = (uint16_t*)carve((size_t)TT * NN * CC * 2);
  uint16_t* encWxT  = (uint16_t*)carve((size_t)CC * DD * 2);
  uint16_t* encWhT  = (uint16_t*)carve((size_t)DD * DD * 2);
  uint16_t* WalphaT = (uint16_t*)carve((size_t)DD * DD * 2);
  uint16_t* decWhT  = (uint16_t*)carve((size_t)DD * DD * 2);
  uint16_t* decWyT  = (uint16_t*)carve((size_t)DD * KK * 2);
  uint16_t* WcT     = (uint16_t*)carve((size_t)(DD + KK) * KK * 2);
  float*    enc_hf  = (float*)   carve((size_t)TT * NN * DD * 4);
  uint16_t* enc_hbf = (uint16_t*)carve((size_t)TT * NN * DD * 2);
  uint16_t* pre_bf  = (uint16_t*)carve((size_t)TT * NN * DD * 2);
  float*    h_f[2]; uint16_t* h_bf[2];
  h_f[0]  = (float*)   carve((size_t)NN * DD * 4);
  h_f[1]  = (float*)   carve((size_t)NN * DD * 4);
  h_bf[0] = (uint16_t*)carve((size_t)NN * DD * 2);
  h_bf[1] = (uint16_t*)carve((size_t)NN * DD * 2);
  float*    e_buf   = (float*)   carve((size_t)TT * NN * 4);
  float*    a_buf   = (float*)   carve((size_t)TT * NN * 4);
  uint16_t* c_bf    = (uint16_t*)carve((size_t)NN * DD * 2);
  uint16_t* yrnn_bf = (uint16_t*)carve((size_t)NN * KK * 2);
  int*      idx     = (int*)     carve((size_t)NN * 4);

  const dim3 gw(32, 4);   // 4 waves per block, one 32x32 macro-tile each

  // ---- precompute: bf16 input + transposed bf16 weights ----
  att_convert_bf<<<(TT * NN * CC + 255) / 256, 256, 0, stream>>>(x, x_bf, TT * NN * CC);
  att_transpose_bf<<<(CC * DD + 255) / 256, 256, 0, stream>>>(enc_Wx, encWxT, CC, DD);
  att_transpose_bf<<<(DD * DD + 255) / 256, 256, 0, stream>>>(enc_Wh, encWhT, DD, DD);
  att_transpose_bf<<<(DD * DD + 255) / 256, 256, 0, stream>>>(W_alpha, WalphaT, DD, DD);
  att_transpose_bf<<<(DD * DD + 255) / 256, 256, 0, stream>>>(dec_Wh, decWhT, DD, DD);
  att_transpose_bf<<<(DD * KK + 255) / 256, 256, 0, stream>>>(dec_Wy, decWyT, DD, KK);
  att_transpose_bf<<<((DD + KK) * KK + 255) / 256, 256, 0, stream>>>(Wc, WcT, DD + KK, KK);
  att_init_idx<<<1, NN, 0, stream>>>(idx);

  // ---- encoder RNN (sequential); 128 macro tiles -> 32 blocks ----
  for (int t = 0; t < TT; t++) {
    const uint16_t* hprev = t ? enc_hbf + (size_t)(t - 1) * NN * DD : enc_hbf;
    att_enc_step<<<32, gw, 0, stream>>>(
        x_bf + (size_t)t * NN * CC, hprev, t ? 1 : 0,
        encWxT, encWhT, enc_bx, enc_bh,
        enc_hf + (size_t)t * NN * DD, enc_hbf + (size_t)t * NN * DD);
  }

  // ---- pre = enc_h @ W_alpha  (16384 x 1024 x 1024); 16384 macro tiles ----
  att_gemm<<<4096, gw, 0, stream>>>(enc_hbf, DD, WalphaT, DD, DD,
                                    nullptr, nullptr, pre_bf, DD, DD / 32);

  // ---- decoder (sequential) ----
  for (int t = 0; t < TT; t++) {
    const float*    hinf = t ? h_f[(t - 1) & 1]  : enc_hf  + (size_t)(TT - 1) * NN * DD;
    const uint16_t* hinb = t ? h_bf[(t - 1) & 1] : enc_hbf + (size_t)(TT - 1) * NN * DD;
    float*    houtf = h_f[t & 1];
    uint16_t* houtb = h_bf[t & 1];

    att_edot<<<TT * NN / 4, 128, 0, stream>>>(pre_bf, hinf, e_buf);
    att_softmax<<<NN, TT, 0, stream>>>(e_buf, a_buf, aij + (size_t)t * TT * NN);
    att_ctx<<<NN * DD / 256, 256, 0, stream>>>(a_buf, enc_hf, c_bf);
    att_dec_hstep<<<32, gw, 0, stream>>>(hinb, decWhT, dec_Wx, idx,
                                         dec_bx, dec_bh, houtf, houtb);
    att_gemm<<<16, gw, 0, stream>>>(houtb, DD, decWyT, DD, DD,
                                    dec_by, nullptr, yrnn_bf, KK, KK / 32);
    att_dec_y<<<16, gw, 0, stream>>>(yrnn_bf, c_bf, WcT, bc,
                                     yout + (size_t)t * NN * KK);
    att_argmax<<<NN, 128, 0, stream>>>(yout + (size_t)t * NN * KK, idx);
  }
}
